// LayernormLinear_84078279786859
// MI455X (gfx1250) — matrix operand order repaired and checked by reading, not compiled, not run on hardware
//
#include <hip/hip_runtime.h>
#include <hip/hip_bf16.h>

typedef __attribute__((ext_vector_type(16))) _Float16 v16h;
typedef __attribute__((ext_vector_type(8)))  float    v8f;

#define LN_EPS 1e-5f
#define LNK 128
#define LNN 128

// ---------------------------------------------------------------------------
// Pre-kernel: build B' = ln_weight ⊙ B as f16 in WMMA B-fragment order, and
// c2[n] = sum_k ln_bias[k] * B[k][n] in fp32.
//
// B-fragment layout for V_WMMA_F32_16X16X32_F16 (wave32, ISA 7.12.2):
//   fragment (kt, nt): lane l holds column n = nt*16 + (l&15);
//   its 16 f16 elements j=0..15 are K = kt*32 + (l>=16 ? 8 : 0) + (j&7) + (j>=8 ? 16 : 0).
// Stored flat: ws[((kt*8 + nt)*32 + lane)*16 + j]  -> each lane's 16 halves
// are 32 contiguous bytes; a wave's fragment is 1 KB contiguous.
// ---------------------------------------------------------------------------
__global__ __launch_bounds__(256) void lnlin_prep_kernel(
    const float* __restrict__ lnw, const float* __restrict__ lnb,
    const float* __restrict__ B, _Float16* __restrict__ wsBfrag,
    float* __restrict__ wsC2)
{
    const int tid = threadIdx.x;
    for (int idx = tid; idx < 4 * 8 * 32 * 16; idx += 256) {
        const int j    = idx & 15;
        const int lane = (idx >> 4) & 31;
        const int nt   = (idx >> 9) & 7;
        const int kt   = idx >> 12;
        const int k    = kt * 32 + ((lane >> 4) << 3) + (j & 7) + ((j >> 3) << 4);
        const int n    = nt * 16 + (lane & 15);
        wsBfrag[idx] = (_Float16)(lnw[k] * B[k * LNN + n]);
    }
    if (tid < LNN) {
        float acc = 0.0f;
        #pragma unroll 8
        for (int k = 0; k < LNK; ++k) acc += lnb[k] * B[k * LNN + tid];
        wsC2[tid] = acc;
    }
}

// ---------------------------------------------------------------------------
// Main fused kernel: 8 waves per WG, each wave owns 16 rows.
// A-fragment layout for 16x32 f16 A (ISA 7.12.2): lane l -> row (l&15);
//   lanes 0-15 hold K = {0..7, 16..23}, lanes 16-31 hold K = {8..15, 24..31}
//   (two contiguous 8-element runs per lane -> float4-friendly).
// ---------------------------------------------------------------------------
__global__ __launch_bounds__(256) void lnlin_main_kernel(
    const float* __restrict__ x, const _Float16* __restrict__ wsBfrag,
    const float* __restrict__ wsC2, float* __restrict__ out)
{
    __shared__ _Float16 sBfrag[4 * 8 * 32 * 16]; // 32 KB, swizzled B' frags
    __shared__ float    sC2[LNN];

    const int tid = threadIdx.x;

    // Coalesced stage of B' fragments (32 KB) and c2 into LDS.
    {
        const uint4* src = (const uint4*)wsBfrag;
        uint4*       dst = (uint4*)sBfrag;
        #pragma unroll
        for (int i = 0; i < 8; ++i) dst[tid + i * 256] = src[tid + i * 256];
        if (tid < LNN) sC2[tid] = wsC2[tid];
    }
    __syncthreads();

    const int wave = tid >> 5;
    const int lane = tid & 31;
    const int hi   = lane >> 4;      // lane half (A/B layout split)
    const int lrow = lane & 15;      // row within 16-row tile / col within 16
    const int kbase = hi * 8;        // this lane's K-chunk base within a 32-tile

    const long rowTile = (long)blockIdx.x * 128 + wave * 16;
    const float* xr = x + (rowTile + lrow) * LNK;

    // Load this lane's 64 x-elements in A-fragment chunk order + LN stats.
    float4 xv[16];
    float s = 0.0f, ss = 0.0f;
    #pragma unroll
    for (int kt = 0; kt < 4; ++kt) {
        const float* p = xr + kt * 32 + kbase;
        xv[kt * 4 + 0] = *(const float4*)(p);
        xv[kt * 4 + 1] = *(const float4*)(p + 4);
        xv[kt * 4 + 2] = *(const float4*)(p + 16);
        xv[kt * 4 + 3] = *(const float4*)(p + 20);
    }
    #pragma unroll
    for (int i = 0; i < 16; ++i) {
        const float4 v = xv[i];
        s  += v.x + v.y + v.z + v.w;
        ss += v.x * v.x + v.y * v.y + v.z * v.z + v.w * v.w;
    }
    // Partner lane (l ^ 16) holds the complementary 64 elements of the row.
    s  += __shfl_xor(s, 16, 32);
    ss += __shfl_xor(ss, 16, 32);
    const float mean = s * (1.0f / 128.0f);
    const float var  = ss * (1.0f / 128.0f) - mean * mean;   // biased variance
    const float rstd = __frsqrt_rn(var + LN_EPS);

    // Build 4 A fragments: x_hat = (x - mean) * rstd, cast to f16.
    v16h aFrag[4];
    #pragma unroll
    for (int kt = 0; kt < 4; ++kt) {
        #pragma unroll
        for (int q = 0; q < 4; ++q) {
            const float4 v = xv[kt * 4 + q];
            const int j = q * 4;
            aFrag[kt][j + 0] = (_Float16)((v.x - mean) * rstd);
            aFrag[kt][j + 1] = (_Float16)((v.y - mean) * rstd);
            aFrag[kt][j + 2] = (_Float16)((v.z - mean) * rstd);
            aFrag[kt][j + 3] = (_Float16)((v.w - mean) * rstd);
        }
    }

    // GEMM: 8 N-tiles, K chained over 4 WMMAs through the accumulator.
    const v16h* sB = (const v16h*)sBfrag;
    float* outBase = out + (rowTile + hi * 8) * LNN + lrow;
    #pragma unroll
    for (int nt = 0; nt < 8; ++nt) {
        v8f acc = {};
        #pragma unroll
        for (int kt = 0; kt < 4; ++kt) {
            const v16h b = sB[(kt * 8 + nt) * 32 + lane];
            acc = __builtin_amdgcn_wmma_f32_16x16x32_f16(
                false, aFrag[kt], false, b, (short)0, acc, false, false);
        }
        const float c2 = sC2[nt * 16 + lrow];
        // D layout: lane l holds col nt*16+(l&15); VGPR v -> row v + 8*(l>>4).
        #pragma unroll
        for (int v = 0; v < 8; ++v) {
            outBase[(long)v * LNN + nt * 16] = acc[v] + c2;
        }
    }
}

extern "C" void kernel_launch(void* const* d_in, const int* in_sizes, int n_in,
                              void* d_out, int out_size, void* d_ws, size_t ws_size,
                              hipStream_t stream) {
    const float* x   = (const float*)d_in[0];
    const float* lnw = (const float*)d_in[1];
    const float* lnb = (const float*)d_in[2];
    const float* B   = (const float*)d_in[3];
    float* out = (float*)d_out;

    _Float16* wsBfrag = (_Float16*)d_ws;                       // 32 KB
    float*    wsC2    = (float*)((char*)d_ws + 32 * 1024);     // 512 B

    const long M = (long)in_sizes[0] / LNK;                    // 524288
    const int  nBlocks = (int)(M / 128);                       // 128 rows per WG

    lnlin_prep_kernel<<<1, 256, 0, stream>>>(lnw, lnb, B, wsBfrag, wsC2);
    lnlin_main_kernel<<<nBlocks, 256, 0, stream>>>(x, wsBfrag, wsC2, out);
}
